// Block_43001212567894
// MI455X (gfx1250) — matrix operand order repaired and checked by reading, not compiled, not run on hardware
//
#include <hip/hip_runtime.h>
#include <hip/hip_bf16.h>
#include <math.h>

#define B_ 4
#define T_ 2048
#define D_ 1024
#define H_ 16
#define HS_ 64

#define USE_ASYNC_LDS 1   // gfx1250 GLOBAL_LOAD_ASYNC_TO_LDS_B128 staging path

typedef unsigned short u16;
typedef __attribute__((ext_vector_type(8)))  unsigned short v8u;
typedef __attribute__((ext_vector_type(16))) unsigned short v16u;
typedef __attribute__((ext_vector_type(16))) __bf16        v16bf;
typedef __attribute__((ext_vector_type(8)))  float         v8f;

static __device__ __forceinline__ u16 f2bfu(float f) {
  unsigned u = __builtin_bit_cast(unsigned, f);
  u = (u + 0x7FFFu + ((u >> 16) & 1u)) >> 16;
  return (u16)u;
}
static __device__ __forceinline__ v16bf mkbf(v8u lo, v8u hi) {
  v16u r;
#pragma unroll
  for (int i = 0; i < 8; ++i) { r[i] = lo[i]; r[i + 8] = hi[i]; }
  return __builtin_bit_cast(v16bf, r);
}
static __device__ __forceinline__ v8f vzero() {
  v8f z;
#pragma unroll
  for (int i = 0; i < 8; ++i) z[i] = 0.0f;
  return z;
}

// ---------------- weight conversion (fp32 [K,N] -> bf16 transposed [N,K]) ----------------
__global__ void cvt_transpose_bf16(const float* __restrict__ in, u16* __restrict__ out,
                                   int N, int K) {
  int i = blockIdx.x * blockDim.x + threadIdx.x;  // over N*K
  if (i >= N * K) return;
  int n = i / K, k = i - n * K;
  out[i] = f2bfu(in[(size_t)k * N + n]);
}

// wq/wk/wv [H,D,HS] -> bf16 [3*H*HS, D] (transposed) so QKV is one GEMM
__global__ void pack_qkv_wT(const float* __restrict__ wq, const float* __restrict__ wk,
                            const float* __restrict__ wv, u16* __restrict__ out) {
  int i = blockIdx.x * blockDim.x + threadIdx.x;  // over 3072*1024
  if (i >= 3 * D_ * D_) return;
  int n = i >> 10;       // output row (QKV column) 0..3071
  int d = i & 1023;      // K index
  int s = n >> 10;       // 0=q,1=k,2=v
  int hk = n & 1023;
  int h = hk >> 6, k = hk & 63;
  const float* w = (s == 0) ? wq : (s == 1) ? wk : wv;
  out[i] = f2bfu(w[((size_t)h * D_ + d) * HS_ + k]);
}

// ---------------- layernorm (fp32 in -> bf16 out) ----------------
__global__ __launch_bounds__(256) void ln_bf16(const float* __restrict__ x,
                                               const float* __restrict__ g,
                                               const float* __restrict__ b,
                                               u16* __restrict__ out) {
  int row = blockIdx.x;
  int tid = threadIdx.x;
  const float4* xr = (const float4*)(x + (size_t)row * D_);
  float4 v = xr[tid];
  float s = v.x + v.y + v.z + v.w;
  float ss = v.x * v.x + v.y * v.y + v.z * v.z + v.w * v.w;
#pragma unroll
  for (int m = 16; m >= 1; m >>= 1) { s += __shfl_xor(s, m); ss += __shfl_xor(ss, m); }
  __shared__ float r1[8], r2[8];
  int wave = tid >> 5, lane = tid & 31;
  if (lane == 0) { r1[wave] = s; r2[wave] = ss; }
  __syncthreads();
  s = 0.0f; ss = 0.0f;
#pragma unroll
  for (int i = 0; i < 8; ++i) { s += r1[i]; ss += r2[i]; }
  float mu = s * (1.0f / D_);
  float var = ss * (1.0f / D_) - mu * mu;
  float rstd = rsqrtf(var + 1e-5f);
  float4 gv = ((const float4*)g)[tid];
  float4 bv = ((const float4*)b)[tid];
  u16* o = out + (size_t)row * D_ + tid * 4;
  o[0] = f2bfu((v.x - mu) * rstd * gv.x + bv.x);
  o[1] = f2bfu((v.y - mu) * rstd * gv.y + bv.y);
  o[2] = f2bfu((v.z - mu) * rstd * gv.z + bv.z);
  o[3] = f2bfu((v.w - mu) * rstd * gv.w + bv.w);
}

// ---------------- WMMA GEMM: out = act(A[M,K]bf16 * Wt[N,K]bf16^T + bias + res) ----------
// Block tile 256x128, BK=32, 8 waves (4x2), wave tile 64x64 => 16 WMMA per k-step from
// 4 A-frag + 4 B-frag loads (1.0 LDS loads/WMMA). Double-buffered LDS staged with gfx1250
// async global->LDS DMA (ASYNCcnt); one barrier per k-step; transfers overlap the WMMAs.
template <int OUT_BF16, int RELU, int HAS_BIAS, int HAS_RES>
__global__ __launch_bounds__(256) void gemm_bf16k(const u16* __restrict__ A,
                                                  const u16* __restrict__ Wt,
                                                  const float* __restrict__ bias,
                                                  const float* __restrict__ res,
                                                  void* __restrict__ outp,
                                                  int M, int N, int K) {
  const int LDT = 40;                 // padded K-stride keeps 16B alignment
  __shared__ u16 sA[2][256 * LDT];    // [m][k], double buffered (40KB)
  __shared__ u16 sB[2][128 * LDT];    // [n][k], double buffered (20KB)
  int bm = blockIdx.y * 256, bn = blockIdx.x * 128;
  int tid = threadIdx.x;
  int wave = tid >> 5, lane = tid & 31;
  int wm = (wave >> 1) * 64, wn = (wave & 1) * 64;
  int lm = lane & 15, lkh = lane >> 4;

  // staging: A -> one 32-elem row per thread; B -> (row = tid/2, 16-elem chunk = tid%2)
  int rB = tid >> 1, ccB = (tid & 1) * 16;
  size_t gaRow = (size_t)(bm + tid) * K * 2;            // byte offsets
  size_t gbRow = ((size_t)(bn + rB) * K + ccB) * 2;

  v8f acc[4][4];
#pragma unroll
  for (int i = 0; i < 4; ++i)
#pragma unroll
    for (int j = 0; j < 4; ++j) acc[i][j] = vzero();

  auto issue = [&](int buf, int k0) {
    unsigned la = (unsigned)(uintptr_t)(void*)&sA[buf][tid * LDT];
    unsigned lb = (unsigned)(uintptr_t)(void*)&sB[buf][rB * LDT + ccB];
    unsigned ga = (unsigned)(gaRow + (size_t)k0 * 2);
    unsigned gb = (unsigned)(gbRow + (size_t)k0 * 2);
#if USE_ASYNC_LDS
    // GVS mode: INST_OFFSET applies to both LDS and global sides (ISA 08 §4.4)
    asm volatile(
        "global_load_async_to_lds_b128 %0, %1, %4\n\t"
        "global_load_async_to_lds_b128 %0, %1, %4 offset:16\n\t"
        "global_load_async_to_lds_b128 %0, %1, %4 offset:32\n\t"
        "global_load_async_to_lds_b128 %0, %1, %4 offset:48\n\t"
        "global_load_async_to_lds_b128 %2, %3, %5\n\t"
        "global_load_async_to_lds_b128 %2, %3, %5 offset:16"
        :
        : "v"(la), "v"(ga), "v"(lb), "v"(gb), "s"(A), "s"(Wt)
        : "memory");
#else
    const u16* pa = (const u16*)((const char*)A + ga);
    const u16* pb = (const u16*)((const char*)Wt + gb);
#pragma unroll
    for (int q = 0; q < 4; ++q)
      *(v8u*)&sA[buf][tid * LDT + q * 8] = *(const v8u*)(pa + q * 8);
    *(v8u*)&sB[buf][rB * LDT + ccB] = *(const v8u*)pb;
    *(v8u*)&sB[buf][rB * LDT + ccB + 8] = *(const v8u*)(pb + 8);
#endif
  };

  int nsteps = K >> 5;
  issue(0, 0);
  for (int i = 0; i < nsteps; ++i) {
#if USE_ASYNC_LDS
    asm volatile("s_wait_asynccnt 0x0" ::: "memory");  // tile i landed in LDS
#endif
    __syncthreads();                                   // visible to all waves
    if (i + 1 < nsteps) issue((i + 1) & 1, (i + 1) * 32);  // overlaps with WMMA below

    const u16* cA = sA[i & 1];
    const u16* cB = sB[i & 1];
    v16bf bf[4];
#pragma unroll
    for (int ni = 0; ni < 4; ++ni) {  // B frag: e0-7 -> k=lkh*8+e, e8-15 -> k=16+lkh*8+e, lane -> n
      int n = wn + ni * 16 + lm;
      v8u lo = *(const v8u*)&cB[n * LDT + lkh * 8];
      v8u hi = *(const v8u*)&cB[n * LDT + 16 + lkh * 8];
      bf[ni] = mkbf(lo, hi);
    }
#pragma unroll
    for (int mi = 0; mi < 4; ++mi) {  // A frag: e0-7 -> k=lkh*8+e, e8-15 -> k=16+lkh*8+e
      int m = wm + mi * 16 + lm;
      v8u lo = *(const v8u*)&cA[m * LDT + lkh * 8];
      v8u hi = *(const v8u*)&cA[m * LDT + 16 + lkh * 8];
      v16bf af = mkbf(lo, hi);
#pragma unroll
      for (int ni = 0; ni < 4; ++ni)
        acc[mi][ni] = __builtin_amdgcn_wmma_f32_16x16x32_bf16(
            false, af, false, bf[ni], (short)0, acc[mi][ni], false, false);
    }
  }

  // epilogue: C layout row = j + 8*(lane>=16), col = lane%16
#pragma unroll
  for (int mi = 0; mi < 4; ++mi) {
#pragma unroll
    for (int ni = 0; ni < 4; ++ni) {
      int n = bn + wn + ni * 16 + lm;
      float bi = HAS_BIAS ? bias[n] : 0.0f;
#pragma unroll
      for (int j = 0; j < 8; ++j) {
        int m = bm + wm + mi * 16 + j + 8 * lkh;
        float v = acc[mi][ni][j] + bi;
        if (HAS_RES) v += res[(size_t)m * N + n];
        if (RELU) v = fmaxf(v, 0.0f);
        if (OUT_BF16) ((u16*)outp)[(size_t)m * N + n] = f2bfu(v);
        else          ((float*)outp)[(size_t)m * N + n] = v;
      }
    }
  }
}

// ---------------- causal flash attention ----------------
// grid = (B*H, T/128); 8 waves; wave handles 16 query rows; s-blocks of 32.
// qkv layout: [B, T, 3, H, HS] bf16 (row stride 3*D).
__global__ __launch_bounds__(256) void attn_kernel(const u16* __restrict__ qkv,
                                                   u16* __restrict__ ctx) {
  const float scale = 0.125f;  // HS^-0.5
  const float NEG = -1e30f;
  int bh = blockIdx.x;
  int b = bh >> 4, h = bh & 15;
  int qbase = blockIdx.y * 128;
  int tid = threadIdx.x;
  int wave = tid >> 5, lane = tid & 31;
  int lm = lane & 15, lkh = lane >> 4;
  int qrow0 = qbase + wave * 16;
  // scalarize the wave-uniform bound -> s_cmp branch, EXEC untouched around WMMA
  int q15 = __builtin_amdgcn_readfirstlane(qrow0 + 15);

  __shared__ u16 ldsK[32 * 64];      // [s][hs] == B^T layout for Q*K^T
  __shared__ u16 ldsV[64 * 32];      // [hs][s] == B layout for P*V
  __shared__ u16 ldsP[8][16 * 32];   // per-wave P staging (C layout -> A layout)

  v16bf qf[2];
  {
    const u16* qp = qkv + ((size_t)(b * T_ + qrow0 + lm) * 3 * D_) + h * HS_;
#pragma unroll
    for (int c = 0; c < 2; ++c) {
      v8u lo = *(const v8u*)(qp + c * 32 + lkh * 8);
      v8u hi = *(const v8u*)(qp + c * 32 + lkh * 8 + 16);
      qf[c] = mkbf(lo, hi);
    }
  }

  float rowm[8], rowl[8];
#pragma unroll
  for (int j = 0; j < 8; ++j) { rowm[j] = NEG; rowl[j] = 0.0f; }
  v8f acc[4];
#pragma unroll
  for (int c = 0; c < 4; ++c) acc[c] = vzero();

  int nblk = qbase / 32 + 4;
  int kr = tid >> 3, kc = (tid & 7) * 8;

  for (int sb = 0; sb < nblk; ++sb) {
    int s0 = sb * 32;
    __syncthreads();
    {  // cooperative K/V tile load (32 rows x 64)
      const u16* kp = qkv + ((size_t)(b * T_ + s0 + kr) * 3 * D_) + D_ + h * HS_ + kc;
      v8u kv = *(const v8u*)kp;
      *(v8u*)&ldsK[kr * 64 + kc] = kv;
      v8u vv = *(const v8u*)(kp + D_);
#pragma unroll
      for (int i = 0; i < 8; ++i) ldsV[(kc + i) * 32 + kr] = vv[i];
    }
    __syncthreads();

    bool active = (s0 <= q15);  // scalar condition
    if (active) {
      v8f sfr[2];
#pragma unroll
      for (int nc = 0; nc < 2; ++nc) {
        v8f s = vzero();
#pragma unroll
        for (int c = 0; c < 2; ++c) {
          v8u lo = *(const v8u*)&ldsK[(nc * 16 + lm) * 64 + c * 32 + lkh * 8];
          v8u hi = *(const v8u*)&ldsK[(nc * 16 + lm) * 64 + c * 32 + lkh * 8 + 16];
          v16bf kf = mkbf(lo, hi);
          s = __builtin_amdgcn_wmma_f32_16x16x32_bf16(false, qf[c], false, kf,
                                                      (short)0, s, false, false);
        }
        sfr[nc] = s;
      }
      float alpha[8];
      int c0 = s0 + lm, c1 = s0 + 16 + lm;
#pragma unroll
      for (int j = 0; j < 8; ++j) {
        int qt = qrow0 + j + 8 * lkh;
        float a0 = (c0 <= qt) ? sfr[0][j] * scale : NEG;
        float a1 = (c1 <= qt) ? sfr[1][j] * scale : NEG;
        float mx = fmaxf(a0, a1);
#pragma unroll
        for (int mk = 8; mk >= 1; mk >>= 1) mx = fmaxf(mx, __shfl_xor(mx, mk));
        float mnew = fmaxf(rowm[j], mx);
        float al = __expf(rowm[j] - mnew);
        float p0 = __expf(a0 - mnew);
        float p1 = __expf(a1 - mnew);
        float rs = p0 + p1;
#pragma unroll
        for (int mk = 8; mk >= 1; mk >>= 1) rs += __shfl_xor(rs, mk);
        rowl[j] = rowl[j] * al + rs;
        rowm[j] = mnew;
        alpha[j] = al;
        int m = j + 8 * lkh;
        ldsP[wave][m * 32 + lm] = f2bfu(p0);
        ldsP[wave][m * 32 + 16 + lm] = f2bfu(p1);
      }
#pragma unroll
      for (int c = 0; c < 4; ++c)
#pragma unroll
        for (int j = 0; j < 8; ++j) acc[c][j] *= alpha[j];
    }
    __syncthreads();
    if (active) {  // ctx += P(16x32) * V(32x64)
      v8u lo = *(const v8u*)&ldsP[wave][lm * 32 + lkh * 8];
      v8u hi = *(const v8u*)&ldsP[wave][lm * 32 + 16 + lkh * 8];
      v16bf pf = mkbf(lo, hi);
#pragma unroll
      for (int c = 0; c < 4; ++c) {
        v8u vlo = *(const v8u*)&ldsV[(c * 16 + lm) * 32 + lkh * 8];
        v8u vhi = *(const v8u*)&ldsV[(c * 16 + lm) * 32 + lkh * 8 + 16];
        v16bf vf = mkbf(vlo, vhi);
        acc[c] = __builtin_amdgcn_wmma_f32_16x16x32_bf16(false, pf, false, vf,
                                                         (short)0, acc[c], false, false);
      }
    }
  }

#pragma unroll
  for (int j = 0; j < 8; ++j) {
    int t = qrow0 + j + 8 * lkh;
    float inv = 1.0f / rowl[j];
#pragma unroll
    for (int c = 0; c < 4; ++c) {
      int col = h * HS_ + c * 16 + lm;
      ctx[(size_t)(b * T_ + t) * D_ + col] = f2bfu(acc[c][j] * inv);
    }
  }
}

// ---------------- launch ----------------
extern "C" void kernel_launch(void* const* d_in, const int* in_sizes, int n_in,
                              void* d_out, int out_size, void* d_ws, size_t ws_size,
                              hipStream_t stream) {
  (void)in_sizes; (void)n_in; (void)out_size; (void)ws_size;
  const float* x      = (const float*)d_in[0];
  const float* ln1_g  = (const float*)d_in[1];
  const float* ln1_b  = (const float*)d_in[2];
  const float* wq     = (const float*)d_in[3];
  const float* wk     = (const float*)d_in[4];
  const float* wv     = (const float*)d_in[5];
  const float* w_proj = (const float*)d_in[6];
  const float* b_proj = (const float*)d_in[7];
  const float* ln2_g  = (const float*)d_in[8];
  const float* ln2_b  = (const float*)d_in[9];
  const float* w1     = (const float*)d_in[10];
  const float* b1     = (const float*)d_in[11];
  const float* w2     = (const float*)d_in[12];
  const float* b2     = (const float*)d_in[13];

  char* ws = (char*)d_ws;
  size_t o = 0;
  u16*   xnb    = (u16*)(ws + o);   o += (size_t)8192 * 1024 * 2;
  u16*   qkvb   = (u16*)(ws + o);   o += (size_t)8192 * 3072 * 2;
  u16*   ctxb   = (u16*)(ws + o);   o += (size_t)8192 * 1024 * 2;
  float* x1     = (float*)(ws + o); o += (size_t)8192 * 1024 * 4;
  u16*   xn2b   = (u16*)(ws + o);   o += (size_t)8192 * 1024 * 2;
  u16*   hb     = (u16*)(ws + o);   o += (size_t)8192 * 4096 * 2;
  u16*   wqkvt  = (u16*)(ws + o);   o += (size_t)3072 * 1024 * 2;
  u16*   wprojt = (u16*)(ws + o);   o += (size_t)1024 * 1024 * 2;
  u16*   w1t    = (u16*)(ws + o);   o += (size_t)4096 * 1024 * 2;
  u16*   w2t    = (u16*)(ws + o);   o += (size_t)1024 * 4096 * 2;

  // weights -> bf16, pre-transposed to [N][K]
  pack_qkv_wT<<<dim3((3072 * 1024 + 255) / 256), 256, 0, stream>>>(wq, wk, wv, wqkvt);
  cvt_transpose_bf16<<<dim3((1024 * 1024 + 255) / 256), 256, 0, stream>>>(w_proj, wprojt, 1024, 1024);
  cvt_transpose_bf16<<<dim3((4096 * 1024 + 255) / 256), 256, 0, stream>>>(w1, w1t, 4096, 1024);
  cvt_transpose_bf16<<<dim3((1024 * 4096 + 255) / 256), 256, 0, stream>>>(w2, w2t, 1024, 4096);

  // LN1 -> QKV GEMM -> attention -> proj(+res) -> LN2 -> FFN1(relu) -> FFN2(+res)
  ln_bf16<<<dim3(8192), 256, 0, stream>>>(x, ln1_g, ln1_b, xnb);
  gemm_bf16k<1, 0, 0, 0><<<dim3(24, 32), 256, 0, stream>>>(xnb, wqkvt, nullptr, nullptr,
                                                           qkvb, 8192, 3072, 1024);
  attn_kernel<<<dim3(64, 16), 256, 0, stream>>>(qkvb, ctxb);
  gemm_bf16k<0, 0, 1, 1><<<dim3(8, 32), 256, 0, stream>>>(ctxb, wprojt, b_proj, x,
                                                          x1, 8192, 1024, 1024);
  ln_bf16<<<dim3(8192), 256, 0, stream>>>(x1, ln2_g, ln2_b, xn2b);
  gemm_bf16k<1, 1, 1, 0><<<dim3(32, 32), 256, 0, stream>>>(xn2b, w1t, b1, nullptr,
                                                           hb, 8192, 4096, 1024);
  gemm_bf16k<0, 0, 1, 1><<<dim3(8, 32), 256, 0, stream>>>(hb, w2t, b2, x1,
                                                          (float*)d_out, 8192, 1024, 4096);
}
